// LAS_32495722561956
// MI455X (gfx1250) — compile-verified, matrix operand order checked by reading, not run on hardware
//
#include <hip/hip_runtime.h>
#include <cstdint>
#include <cstddef>

// ---------------------------------------------------------------------------
// LAS forward for MI455X (gfx1250, wave32, WMMA).
// All matmuls run through v_wmma_f32_16x16x32_bf16 (bf16 A/B, f32 accum).
// Batch B=16 == WMMA M tile; each wave owns one 16x16 C tile.
// A (activation) tiles are staged to LDS once per workgroup with
// global_load_async_to_lds_b128 (ASYNCcnt) and read back as WMMA fragments
// via ds_load_b128; W (weight) tiles stream from global/L2.
// ---------------------------------------------------------------------------

typedef __attribute__((ext_vector_type(16))) __bf16 v16bf;
typedef __attribute__((ext_vector_type(8)))  __bf16 v8bf;
typedef __attribute__((ext_vector_type(8)))  float  v8f;
typedef __attribute__((address_space(3)))    __bf16 lds_bf16_t;

union BFrag { v16bf v; v8bf h[2]; };

// One async 16-byte global->LDS copy per lane (CDNA5 GLOBAL_LOAD_ASYNC_TO_LDS_B128).
__device__ __forceinline__ void async_copy16(const __bf16* g, __bf16* l) {
    unsigned loff = (unsigned)(uintptr_t)(lds_bf16_t*)l;   // LDS byte offset
    asm volatile("global_load_async_to_lds_b128 %0, %1, off"
                 :: "v"(loff), "v"(g) : "memory");
}

// ---------------------------------------------------------------------------
// Generic WMMA GEMM:  C[M,N] = A[M,K] * W[N,K]^T (+ A2[M,K2]*W2[N,K2]^T)
//                              (+ Cinit[M,N]) (+ bias[N])
// A, W bf16 row-major; C f32 row-major. M,N multiples of 16; K,K2 of 32.
// Grid: x = ceil((N/16)/8) blocks of 8 waves, y = M/16.
// Dynamic LDS: 16*(K+8) (+ 16*(K2+8)) bf16 (+8 pad breaks bank conflicts).
// Fragment layout per CDNA5 ISA 7.12.2:
//   lanes 0-15  hold K-halves {0..7, 16..23}
//   lanes 16-31 hold K-halves {8..15, 24..31}
//   C tile: VGPR r -> row mbase + (lane>=16 ? 8 : 0) + r, col nbase + (lane&15)
// ---------------------------------------------------------------------------
__global__ __launch_bounds__(256) void gemm_wmma_bf16(
    const __bf16* __restrict__ A,  int lda,
    const __bf16* __restrict__ W,  int ldw,  int K,
    const __bf16* __restrict__ A2, int lda2,
    const __bf16* __restrict__ W2, int ldw2, int K2,
    const float*  __restrict__ Cinit,
    const float*  __restrict__ bias,
    float* __restrict__ C, int N)
{
    extern __shared__ __bf16 smem[];
    const int mbase = blockIdx.y << 4;
    const int ldp   = K + 8;
    const int ldp2  = K2 + 8;
    __bf16* smA  = smem;
    __bf16* smA2 = smem + 16 * ldp;

    // ---- stage A (and A2) tiles into LDS via async copies ----
    {
        int perRow = K >> 3;                 // 16B chunks per row
        int chunks = perRow << 4;            // 16 rows
        for (int c = threadIdx.x; c < chunks; c += blockDim.x) {
            int r = c / perRow, j = (c - r * perRow) << 3;
            async_copy16(A + (size_t)(mbase + r) * lda + j, smA + r * ldp + j);
        }
        if (A2) {
            int perRow2 = K2 >> 3;
            int chunks2 = perRow2 << 4;
            for (int c = threadIdx.x; c < chunks2; c += blockDim.x) {
                int r = c / perRow2, j = (c - r * perRow2) << 3;
                async_copy16(A2 + (size_t)(mbase + r) * lda2 + j, smA2 + r * ldp2 + j);
            }
        }
        asm volatile("s_wait_asynccnt 0" ::: "memory");
    }
    __syncthreads();

    const int wave  = threadIdx.x >> 5;
    const int lane  = threadIdx.x & 31;
    const int ntile = blockIdx.x * 8 + wave;
    const int ntiles = N >> 4;
    if (ntile >= ntiles) return;             // wave-uniform guard (after barrier)

    const int nbase = ntile << 4;
    const int l15   = lane & 15;
    const int kh    = (lane >> 4) << 3;      // 0 or 8

    const __bf16* arow = smA + l15 * ldp;                          // LDS
    const __bf16* wrow = W + (size_t)(nbase + l15) * ldw;          // global/L2

    const int ccol  = nbase + l15;
    const int crow0 = mbase + ((lane >> 4) << 3);

    v8f acc;
    if (Cinit) {
        #pragma unroll
        for (int r = 0; r < 8; ++r)
            acc[r] = Cinit[(size_t)(crow0 + r) * N + ccol];
    } else {
        #pragma unroll
        for (int r = 0; r < 8; ++r) acc[r] = 0.0f;
    }

    for (int k = 0; k < K; k += 32) {
        BFrag a, b;
        a.h[0] = *(const v8bf*)(arow + k + kh);        // ds_load_b128
        a.h[1] = *(const v8bf*)(arow + k + 16 + kh);
        b.h[0] = *(const v8bf*)(wrow + k + kh);        // global_load_b128
        b.h[1] = *(const v8bf*)(wrow + k + 16 + kh);
        __builtin_prefetch(wrow + k + 128, 0, 3);      // near-scope weight prefetch
        acc = __builtin_amdgcn_wmma_f32_16x16x32_bf16(
                false, a.v, false, b.v, (short)0, acc, false, false);
    }
    if (A2) {
        const __bf16* arow2 = smA2 + l15 * ldp2;
        const __bf16* wrow2 = W2 + (size_t)(nbase + l15) * ldw2;
        for (int k = 0; k < K2; k += 32) {
            BFrag a, b;
            a.h[0] = *(const v8bf*)(arow2 + k + kh);
            a.h[1] = *(const v8bf*)(arow2 + k + 16 + kh);
            b.h[0] = *(const v8bf*)(wrow2 + k + kh);
            b.h[1] = *(const v8bf*)(wrow2 + k + 16 + kh);
            __builtin_prefetch(wrow2 + k + 128, 0, 3);
            acc = __builtin_amdgcn_wmma_f32_16x16x32_bf16(
                    false, a.v, false, b.v, (short)0, acc, false, false);
        }
    }
    if (bias) {
        float bv = bias[ccol];
        #pragma unroll
        for (int r = 0; r < 8; ++r) acc[r] += bv;
    }
    #pragma unroll
    for (int r = 0; r < 8; ++r)
        C[(size_t)(crow0 + r) * N + ccol] = acc[r];
}

// --------------------------- small utility kernels -------------------------
__global__ void f32_to_bf16(const float* __restrict__ s, __bf16* __restrict__ d, size_t n) {
    size_t i = (size_t)blockIdx.x * blockDim.x + threadIdx.x;
    size_t st = (size_t)gridDim.x * blockDim.x;
    for (; i < n; i += st) d[i] = (__bf16)s[i];
}
__global__ void zero_f32(float* p, int n) {
    int i = blockIdx.x * blockDim.x + threadIdx.x;
    if (i < n) p[i] = 0.0f;
}
__global__ void zero_bf16(__bf16* p, int n) {
    int i = blockIdx.x * blockDim.x + threadIdx.x;
    if (i < n) p[i] = (__bf16)0.0f;
}

// --------------------------- conv + BN frontend ----------------------------
// input view: in[b, ch, t, f] = spec[b, ch*40+f, t]  (B=16, SPEC=80, TS=512)
__global__ void conv1_kernel(const float* __restrict__ spec, const float* __restrict__ w,
                             const float* __restrict__ bias, float* __restrict__ out) {
    int idx = blockIdx.x * blockDim.x + threadIdx.x;
    if (idx >= 16 * 32 * 256 * 20) return;
    int x = idx % 20, y = (idx / 20) % 256, o = (idx / (20 * 256)) % 32, b = idx / (20 * 256 * 32);
    float acc = bias[o];
    for (int i = 0; i < 2; ++i)
        for (int ky = 0; ky < 3; ++ky) {
            int hy = 2 * y + ky - 1;
            if (hy < 0 || hy >= 512) continue;
            for (int kx = 0; kx < 3; ++kx) {
                int hx = 2 * x + kx - 1;
                if (hx < 0 || hx >= 40) continue;
                float v = spec[(size_t)b * 80 * 512 + (size_t)(i * 40 + hx) * 512 + hy];
                acc += v * w[((o * 2 + i) * 3 + ky) * 3 + kx];
            }
        }
    out[idx] = acc;
}

__global__ void conv2_kernel(const float* __restrict__ in, const float* __restrict__ w,
                             const float* __restrict__ bias, float* __restrict__ out) {
    int idx = blockIdx.x * blockDim.x + threadIdx.x;
    if (idx >= 16 * 32 * 128 * 10) return;
    int x = idx % 10, y = (idx / 10) % 128, o = (idx / 1280) % 32, b = idx / 40960;
    float acc = bias[o];
    for (int i = 0; i < 32; ++i)
        for (int ky = 0; ky < 3; ++ky) {
            int hy = 2 * y + ky - 1;
            if (hy < 0 || hy >= 256) continue;
            for (int kx = 0; kx < 3; ++kx) {
                int hx = 2 * x + kx - 1;
                if (hx < 0 || hx >= 20) continue;
                acc += in[((size_t)(b * 32 + i) * 256 + hy) * 20 + hx] *
                       w[((o * 32 + i) * 3 + ky) * 3 + kx];
            }
        }
    out[idx] = acc;
}

// batch-norm (training mode): stats over (N,H,W) per channel
__global__ void bn_stats(const float* __restrict__ x, float* __restrict__ stats,
                         int C, int HW, int B) {
    int c = blockIdx.x;
    int per = B * HW;
    float s = 0.f, s2 = 0.f;
    for (int e = threadIdx.x; e < per; e += blockDim.x) {
        int b = e / HW, r = e % HW;
        float v = x[((size_t)(b * C + c)) * HW + r];
        s += v; s2 += v * v;
    }
    __shared__ float rs[256], rq[256];
    rs[threadIdx.x] = s; rq[threadIdx.x] = s2; __syncthreads();
    for (int o = 128; o > 0; o >>= 1) {
        if ((int)threadIdx.x < o) { rs[threadIdx.x] += rs[threadIdx.x + o]; rq[threadIdx.x] += rq[threadIdx.x + o]; }
        __syncthreads();
    }
    if (threadIdx.x == 0) {
        float m = rs[0] / per;
        stats[c] = m;
        stats[C + c] = rq[0] / per - m * m;
    }
}

__global__ void bn_apply_relu(float* __restrict__ x, const float* __restrict__ stats,
                              const float* __restrict__ g, const float* __restrict__ bta,
                              int C, int HW, int total) {
    int idx = blockIdx.x * blockDim.x + threadIdx.x;
    if (idx >= total) return;
    int c = (idx / HW) % C;
    float y = (x[idx] - stats[c]) * rsqrtf(stats[C + c] + 1e-5f) * g[c] + bta[c];
    x[idx] = fmaxf(y, 0.f);
}

// BN2 + relu, emitting X0 bf16 [T*B=2048, 320]: X0[(t*16+b)][c*10+x]
__global__ void bn2_apply_to_x0(const float* __restrict__ x, const float* __restrict__ stats,
                                const float* __restrict__ g, const float* __restrict__ bta,
                                __bf16* __restrict__ X0) {
    int idx = blockIdx.x * blockDim.x + threadIdx.x;
    if (idx >= 16 * 32 * 128 * 10) return;
    int xx = idx % 10, t = (idx / 10) % 128, c = (idx / 1280) % 32, b = idx / 40960;
    float y = (x[idx] - stats[c]) * rsqrtf(stats[32 + c] + 1e-5f) * g[c] + bta[c];
    y = fmaxf(y, 0.f);
    X0[(size_t)(t * 16 + b) * 320 + c * 10 + xx] = (__bf16)y;
}

// --------------------------- LSTM pointwise cell ---------------------------
// gates[16,4096] = [i|f|g|o] blocks of 1024 (torch gate order)
__global__ void lstm_cell(const float* __restrict__ gates, float* __restrict__ h,
                          float* __restrict__ c, __bf16* __restrict__ hbf,
                          __bf16* __restrict__ seq, int seqld, int coloff, int t) {
    int idx = blockIdx.x * blockDim.x + threadIdx.x;
    if (idx >= 16 * 1024) return;
    int b = idx >> 10, j = idx & 1023;
    const float* gb = gates + (size_t)b * 4096;
    float gi = gb[j], gf = gb[1024 + j], gg = gb[2048 + j], go = gb[3072 + j];
    float si = 1.f / (1.f + __expf(-gi));
    float sf = 1.f / (1.f + __expf(-gf));
    float so = 1.f / (1.f + __expf(-go));
    float cn = sf * c[idx] + si * tanhf(gg);
    float hn = so * tanhf(cn);
    c[idx] = cn; h[idx] = hn; hbf[idx] = (__bf16)hn;
    if (seq) seq[(size_t)(t * 16 + b) * seqld + coloff + j] = (__bf16)hn;
}

// --------------------------- attention (per batch) -------------------------
// scores[t] = <h_proj[t*16+b], s_proj[b]>; softmax over t; context -> xcat[:,256:]
__global__ void attention_kernel(const float* __restrict__ hproj,
                                 const float* __restrict__ sproj,
                                 float* __restrict__ xcat) {
    int b = blockIdx.x, t = threadIdx.x;      // 128 threads
    __shared__ float sc[128], red[128];
    const float4* sp = (const float4*)(sproj + (size_t)b * 2048);
    const float4* hp = (const float4*)(hproj + (size_t)(t * 16 + b) * 2048);
    float d = 0.f;
    for (int j = 0; j < 512; ++j) {
        float4 a = hp[j], q = sp[j];
        d += a.x * q.x + a.y * q.y + a.z * q.z + a.w * q.w;
    }
    sc[t] = d; red[t] = d; __syncthreads();
    for (int s = 64; s > 0; s >>= 1) { if (t < s) red[t] = fmaxf(red[t], red[t + s]); __syncthreads(); }
    float mx = red[0]; __syncthreads();
    float e = __expf(sc[t] - mx);
    sc[t] = e; red[t] = e; __syncthreads();
    for (int s = 64; s > 0; s >>= 1) { if (t < s) red[t] += red[t + s]; __syncthreads(); }
    float inv = 1.f / red[0]; __syncthreads();
    sc[t] *= inv; __syncthreads();
    int e0 = t * 16;
    float ctx[16];
    #pragma unroll
    for (int i = 0; i < 16; ++i) ctx[i] = 0.f;
    for (int tt = 0; tt < 128; ++tt) {
        float a = sc[tt];
        const float* hh = hproj + (size_t)(tt * 16 + b) * 2048 + e0;
        #pragma unroll
        for (int i = 0; i < 16; ++i) ctx[i] += a * hh[i];
    }
    #pragma unroll
    for (int i = 0; i < 16; ++i) xcat[(size_t)b * 2304 + 256 + e0 + i] = ctx[i];
}

// teacher-forced embedding: step t uses emb[token_{t-1}] (emb[0] at t=0)
__global__ void embed_kernel(const int* __restrict__ text, const float* __restrict__ emb,
                             float* __restrict__ xcat, int t) {
    int idx = blockIdx.x * blockDim.x + threadIdx.x;
    if (idx >= 16 * 256) return;
    int b = idx >> 8, e = idx & 255;
    int tok = (t == 0) ? 0 : text[b * 128 + (t - 1)];
    xcat[(size_t)b * 2304 + e] = emb[tok * 256 + e];
}

__global__ void logsoftmax_kernel(const float* __restrict__ logits, float* __restrict__ out, int t) {
    int b = blockIdx.x, v = threadIdx.x;      // 64 threads
    __shared__ float red[64];
    float x = logits[b * 64 + v];
    red[v] = x; __syncthreads();
    for (int s = 32; s > 0; s >>= 1) { if (v < s) red[v] = fmaxf(red[v], red[v + s]); __syncthreads(); }
    float mx = red[0]; __syncthreads();
    red[v] = __expf(x - mx); __syncthreads();
    for (int s = 32; s > 0; s >>= 1) { if (v < s) red[v] += red[v + s]; __syncthreads(); }
    float lse = mx + logf(red[0]);
    out[((size_t)b * 128 + t) * 64 + v] = x - lse;
}

// outputs 0 and 2 are int32 pass-throughs; write raw bit patterns
__global__ void copy_int_outputs(const int* __restrict__ text, const int* __restrict__ lens,
                                 int* __restrict__ out) {
    int i = blockIdx.x * blockDim.x + threadIdx.x;
    if (i < 2048) out[i] = text[i];
    else if (i < 2064) out[2048 + 16 * 128 * 64 + (i - 2048)] = lens[i - 2048];
}

// ---------------------------------------------------------------------------
extern "C" void kernel_launch(void* const* d_in, const int* in_sizes, int n_in,
                              void* d_out, int out_size, void* d_ws, size_t ws_size,
                              hipStream_t stream) {
    (void)in_sizes; (void)n_in; (void)out_size; (void)ws_size;

    const float* spec    = (const float*)d_in[0];
    const int*   text    = (const int*)d_in[1];
    const int*   lens    = (const int*)d_in[2];
    const float* conv1_w = (const float*)d_in[4];
    const float* conv1_b = (const float*)d_in[5];
    const float* bn1_g   = (const float*)d_in[6];
    const float* bn1_b   = (const float*)d_in[7];
    const float* conv2_w = (const float*)d_in[8];
    const float* conv2_b = (const float*)d_in[9];
    const float* bn2_g   = (const float*)d_in[10];
    const float* bn2_b   = (const float*)d_in[11];
    // enc layer l dir d: d_in[12 + 6*l + 3*d + {0:Wih,1:Whh,2:b}]
    const float* att_bh  = (const float*)d_in[37];
    const float* att_bs  = (const float*)d_in[39];
    const float* dec_b0  = (const float*)d_in[42];
    const float* dec_b1  = (const float*)d_in[45];
    const float* ct_b    = (const float*)d_in[47];
    const float* embW    = (const float*)d_in[48];

    // ---- carve workspace (256B aligned) ----
    char* wsp = (char*)d_ws;
    auto alloc = [&](size_t bytes) -> void* {
        wsp = (char*)(((uintptr_t)wsp + 255) & ~(uintptr_t)255);
        void* r = (void*)wsp;
        wsp += bytes;
        return r;
    };
    const int encDin[4] = {320, 2048, 2048, 2048};
    __bf16* encWih[4][2]; __bf16* encWhh[4][2];
    for (int l = 0; l < 4; ++l)
        for (int d = 0; d < 2; ++d) {
            encWih[l][d] = (__bf16*)alloc((size_t)4096 * encDin[l] * 2);
            encWhh[l][d] = (__bf16*)alloc((size_t)4096 * 1024 * 2);
        }
    __bf16* attWh = (__bf16*)alloc((size_t)2048 * 2048 * 2);
    __bf16* attWs = (__bf16*)alloc((size_t)2048 * 1024 * 2);
    __bf16* dWih0 = (__bf16*)alloc((size_t)4096 * 2304 * 2);
    __bf16* dWhh0 = (__bf16*)alloc((size_t)4096 * 1024 * 2);
    __bf16* dWih1 = (__bf16*)alloc((size_t)4096 * 1024 * 2);
    __bf16* dWhh1 = (__bf16*)alloc((size_t)4096 * 1024 * 2);
    __bf16* ctW   = (__bf16*)alloc((size_t)64 * 1024 * 2);
    __bf16* X0bf  = (__bf16*)alloc((size_t)2048 * 320 * 2);
    __bf16* XA    = (__bf16*)alloc((size_t)2048 * 2048 * 2);
    __bf16* XB    = (__bf16*)alloc((size_t)2048 * 2048 * 2);
    __bf16* hbf   = (__bf16*)alloc((size_t)16 * 1024 * 2);
    __bf16* h0bf  = (__bf16*)alloc((size_t)16 * 1024 * 2);
    __bf16* h1bf  = (__bf16*)alloc((size_t)16 * 1024 * 2);
    __bf16* xcatbf= (__bf16*)alloc((size_t)16 * 2304 * 2);
    float* conv1o = (float*)alloc((size_t)2621440 * 4);
    float* conv2o = (float*)alloc((size_t)655360 * 4);
    float* stats1 = (float*)alloc(64 * 4);
    float* stats2 = (float*)alloc(64 * 4);
    float* XW     = (float*)alloc((size_t)2048 * 4096 * 4);
    float* hproj  = (float*)alloc((size_t)2048 * 2048 * 4);
    float* gates  = (float*)alloc((size_t)16 * 4096 * 4);
    float* hf32   = (float*)alloc((size_t)16 * 1024 * 4);
    float* cf32   = (float*)alloc((size_t)16 * 1024 * 4);
    float* h0     = (float*)alloc((size_t)16 * 1024 * 4);
    float* c0     = (float*)alloc((size_t)16 * 1024 * 4);
    float* h1     = (float*)alloc((size_t)16 * 1024 * 4);
    float* c1     = (float*)alloc((size_t)16 * 1024 * 4);
    float* sproj  = (float*)alloc((size_t)16 * 2048 * 4);
    float* xcat   = (float*)alloc((size_t)16 * 2304 * 4);
    float* logits = (float*)alloc((size_t)16 * 64 * 4);

    auto cvt = [&](const void* s, __bf16* d, size_t n) {
        int blocks = (int)((n + 255) / 256);
        if (blocks > 4096) blocks = 4096;
        f32_to_bf16<<<blocks, 256, 0, stream>>>((const float*)s, d, n);
    };
    // ---- weights -> bf16 (every call; deterministic) ----
    for (int l = 0; l < 4; ++l)
        for (int d = 0; d < 2; ++d) {
            int base = 12 + 6 * l + 3 * d;
            cvt(d_in[base + 0], encWih[l][d], (size_t)4096 * encDin[l]);
            cvt(d_in[base + 1], encWhh[l][d], (size_t)4096 * 1024);
        }
    cvt(d_in[36], attWh, (size_t)2048 * 2048);
    cvt(d_in[38], attWs, (size_t)2048 * 1024);
    cvt(d_in[40], dWih0, (size_t)4096 * 2304);
    cvt(d_in[41], dWhh0, (size_t)4096 * 1024);
    cvt(d_in[43], dWih1, (size_t)4096 * 1024);
    cvt(d_in[44], dWhh1, (size_t)4096 * 1024);
    cvt(d_in[46], ctW,   (size_t)64 * 1024);

    auto gemm = [&](const __bf16* A, int lda, const __bf16* W, int ldw, int K,
                    const __bf16* A2, int lda2, const __bf16* W2, int ldw2, int K2,
                    const float* Ci, const float* bias, float* C, int M, int N) {
        dim3 g((N / 16 + 7) / 8, M / 16);
        size_t shmem = (size_t)2 * 16 * ((K + 8) + (A2 ? (K2 + 8) : 0));
        gemm_wmma_bf16<<<g, 256, shmem, stream>>>(A, lda, W, ldw, K, A2, lda2, W2, ldw2, K2,
                                                  Ci, bias, C, N);
    };

    // ---- conv frontend ----
    conv1_kernel<<<(2621440 + 255) / 256, 256, 0, stream>>>(spec, conv1_w, conv1_b, conv1o);
    bn_stats<<<32, 256, 0, stream>>>(conv1o, stats1, 32, 256 * 20, 16);
    bn_apply_relu<<<(2621440 + 255) / 256, 256, 0, stream>>>(conv1o, stats1, bn1_g, bn1_b,
                                                             32, 256 * 20, 2621440);
    conv2_kernel<<<(655360 + 255) / 256, 256, 0, stream>>>(conv1o, conv2_w, conv2_b, conv2o);
    bn_stats<<<32, 256, 0, stream>>>(conv2o, stats2, 32, 128 * 10, 16);
    bn2_apply_to_x0<<<(655360 + 255) / 256, 256, 0, stream>>>(conv2o, stats2, bn2_g, bn2_b, X0bf);

    // ---- encoder: 4 bidirectional LSTM layers ----
    const __bf16* Xin = X0bf;
    __bf16* Xout = XA;
    int din = 320;
    for (int l = 0; l < 4; ++l) {
        for (int d = 0; d < 2; ++d) {
            const float* bptr = (const float*)d_in[12 + 6 * l + 3 * d + 2];
            // hoisted input projection: XW[T*B,4096] = X * Wih^T + b   (big WMMA GEMM)
            gemm(Xin, din, encWih[l][d], din, din,
                 nullptr, 0, nullptr, 0, 0, nullptr, bptr, XW, 2048, 4096);
            zero_f32 <<<64, 256, 0, stream>>>(hf32, 16384);
            zero_f32 <<<64, 256, 0, stream>>>(cf32, 16384);
            zero_bf16<<<64, 256, 0, stream>>>(hbf, 16384);
            for (int s = 0; s < 128; ++s) {
                int t = d ? (127 - s) : s;
                // gates = XW[t] + h * Whh^T   (Whh bf16 stays hot in L2)
                gemm(hbf, 1024, encWhh[l][d], 1024, 1024,
                     nullptr, 0, nullptr, 0, 0,
                     XW + (size_t)t * 16 * 4096, nullptr, gates, 16, 4096);
                lstm_cell<<<64, 256, 0, stream>>>(gates, hf32, cf32, hbf,
                                                  Xout, 2048, d * 1024, t);
            }
        }
        Xin = Xout;
        Xout = (Xout == XA) ? XB : XA;
        din = 2048;
    }

    // ---- attention keys/values: h_proj = enc * Wh^T + bh ----
    gemm(Xin, 2048, attWh, 2048, 2048,
         nullptr, 0, nullptr, 0, 0, nullptr, att_bh, hproj, 2048, 2048);

    // ---- decoder: 128 teacher-forced steps ----
    zero_f32 <<<64, 256, 0, stream>>>(h0, 16384);
    zero_f32 <<<64, 256, 0, stream>>>(c0, 16384);
    zero_f32 <<<64, 256, 0, stream>>>(h1, 16384);
    zero_f32 <<<64, 256, 0, stream>>>(c1, 16384);
    zero_bf16<<<64, 256, 0, stream>>>(h0bf, 16384);
    zero_bf16<<<64, 256, 0, stream>>>(h1bf, 16384);

    float* outLogp = (float*)d_out + 2048;   // after int32 tokens
    for (int t = 0; t < 128; ++t) {
        // query projection from carry h0
        gemm(h0bf, 1024, attWs, 1024, 1024,
             nullptr, 0, nullptr, 0, 0, nullptr, att_bs, sproj, 16, 2048);
        embed_kernel<<<16, 256, 0, stream>>>(text, embW, xcat, t);
        attention_kernel<<<16, 128, 0, stream>>>(hproj, sproj, xcat);
        cvt(xcat, xcatbf, (size_t)16 * 2304);
        // dec LSTM layer 0: gates = xcat*Wih0^T + h0*Whh0^T + b0 (fused dual GEMM)
        gemm(xcatbf, 2304, dWih0, 2304, 2304,
             h0bf, 1024, dWhh0, 1024, 1024, nullptr, dec_b0, gates, 16, 4096);
        lstm_cell<<<64, 256, 0, stream>>>(gates, h0, c0, h0bf, nullptr, 0, 0, 0);
        // dec LSTM layer 1
        gemm(h0bf, 1024, dWih1, 1024, 1024,
             h1bf, 1024, dWhh1, 1024, 1024, nullptr, dec_b1, gates, 16, 4096);
        lstm_cell<<<64, 256, 0, stream>>>(gates, h1, c1, h1bf, nullptr, 0, 0, 0);
        // logits + log_softmax
        gemm(h1bf, 1024, ctW, 1024, 1024,
             nullptr, 0, nullptr, 0, 0, nullptr, ct_b, logits, 16, 64);
        logsoftmax_kernel<<<16, 64, 0, stream>>>(logits, outLogp, t);
    }

    copy_int_outputs<<<(2064 + 255) / 256, 256, 0, stream>>>(text, lens, (int*)d_out);
}
